// Seq2SeqAE_52965536694669
// MI455X (gfx1250) — matrix operand order repaired and checked by reading, not compiled, not run on hardware
//
#include <hip/hip_runtime.h>

// ---------------------------------------------------------------------------
// Persistent fused Seq2Seq GRU (encoder + decoder, teacher forcing) for
// MI455X / gfx1250.  One kernel launch runs all 256 recurrent steps.
//   - batch is partitioned: 16 rows per wave, 4 waves per block, 512 blocks
//   - h @ Whh^T, x @ Wih^T, h @ outW^T all via v_wmma_f32_16x16x32_bf16
//   - Whh kept in LDS pre-swizzled into WMMA B-fragment layout
//   - h kept in registers (f32, C-layout) + bf16 LDS staging tile that is
//     re-read each step as the WMMA A-fragment
// ---------------------------------------------------------------------------

#define T_STEPS 128
#define BATCH   32768
#define F_DIM   8
#define H_DIM   64
#define NTILE   12          // 3H / 16 output tiles
#define WAVES   4
#define ROWS_W  16          // batch rows per wave

typedef __attribute__((ext_vector_type(16))) __bf16       v16bf;
typedef __attribute__((ext_vector_type(8)))  float        v8f;
typedef __attribute__((ext_vector_type(8)))  unsigned int v8u;

// f32 -> bf16 (round-to-nearest-even), packed pair in one dword
__device__ __forceinline__ unsigned int pk_bf16(float a, float b) {
    unsigned int ua = __builtin_bit_cast(unsigned int, a);
    unsigned int ub = __builtin_bit_cast(unsigned int, b);
    ua += 0x7FFFu + ((ua >> 16) & 1u);
    ub += 0x7FFFu + ((ub >> 16) & 1u);
    return (ua >> 16) | (ub & 0xFFFF0000u);
}
__device__ __forceinline__ unsigned short f2bf(float a) {
    unsigned int ua = __builtin_bit_cast(unsigned int, a);
    ua += 0x7FFFu + ((ua >> 16) & 1u);
    return (unsigned short)(ua >> 16);
}
__device__ __forceinline__ v16bf mk_frag(unsigned int a0, unsigned int a1,
                                         unsigned int a2, unsigned int a3,
                                         unsigned int a4, unsigned int a5,
                                         unsigned int a6, unsigned int a7) {
    v8u t = {a0, a1, a2, a3, a4, a5, a6, a7};
    return __builtin_bit_cast(v16bf, t);
}
__device__ __forceinline__ v8f wmma_bf16(v16bf a, v16bf b, v8f c) {
    // (neg_a, A, neg_b, B, c_mod, C, reuse_a, reuse_b)
    return __builtin_amdgcn_wmma_f32_16x16x32_bf16(false, a, false, b,
                                                   (short)0, c, false, false);
}
__device__ __forceinline__ float fsigmoid(float x) {
    return __builtin_amdgcn_rcpf(1.0f + __expf(-x));
}
__device__ __forceinline__ float ftanh(float x) {
    float e = __expf(2.0f * x);
    return (e - 1.0f) * __builtin_amdgcn_rcpf(e + 1.0f);
}

__global__ __launch_bounds__(128, 1)
void seq2seq_gru_persistent(
    const float* __restrict__ inputs, const float* __restrict__ target,
    const float* __restrict__ eWih,  const float* __restrict__ eWhh,
    const float* __restrict__ eBih,  const float* __restrict__ eBhh,
    const float* __restrict__ dWih,  const float* __restrict__ dWhh,
    const float* __restrict__ dBih,  const float* __restrict__ dBhh,
    const float* __restrict__ oW,    const float* __restrict__ oB,
    float* __restrict__ out)
{
    // Whh as bf16 B-fragment image: [nt][ks][lane][8 dwords]  (24 KB)
    __shared__ unsigned int   whhf[NTILE * 2 * 32 * 8];
    // per-wave bf16 h staging tile, row-major [16 rows][64 cols]  (8 KB)
    __shared__ unsigned short stage[WAVES][ROWS_W][H_DIM];

    const int tid   = threadIdx.x;
    const int wave  = tid >> 5;
    const int lane  = tid & 31;
    const int laneN = lane & 15;     // C-layout column within a 16-wide tile
    const int laneH = lane >> 4;     // 0: rows 0-7 of C tile, 1: rows 8-15
    const int b0    = blockIdx.x * (WAVES * ROWS_W) + wave * ROWS_W;

    // persistent per-wave state
    v8f          hstate[4];          // h, f32, C-layout (4 col-tiles x 8 rows)
    unsigned int wih[NTILE][4];      // Wih B-fragments (K padded 8->32)
    float        b_r[4], b_z[4], b_in[4], b_hn[4];

    // ---- convert Whh into WMMA B-fragment layout in LDS -------------------
    auto fill_whh = [&](const float* Whh) {
        __syncthreads();
        for (int d = tid; d < NTILE * 2 * 32 * 8; d += blockDim.x) {
            int v  = d & 7;
            int ln = (d >> 3) & 31;
            int ks = (d >> 8) & 1;
            int nt = d >> 9;
            int k0 = 32 * ks + 16 * (ln >> 4) + 2 * v;   // B: lanes<16 K 0-15, lanes>=16 K 16-31
            int cc = 16 * nt + (ln & 15);                // output unit (column)
            whhf[d] = pk_bf16(Whh[cc * H_DIM + k0], Whh[cc * H_DIM + k0 + 1]);
        }
        __syncthreads();
    };

    auto load_gate_consts = [&](const float* Wih, const float* bih, const float* bhh) {
        #pragma unroll
        for (int nt = 0; nt < NTILE; ++nt) {
            #pragma unroll
            for (int v = 0; v < 4; ++v) {
                unsigned int w = 0u;
                if (lane < 16) {                          // K>=16 half is all zero
                    int cc = 16 * nt + lane;
                    w = pk_bf16(Wih[cc * F_DIM + 2 * v], Wih[cc * F_DIM + 2 * v + 1]);
                }
                wih[nt][v] = w;
            }
        }
        #pragma unroll
        for (int jg = 0; jg < 4; ++jg) {
            int cc   = 16 * jg + laneN;
            b_r[jg]  = bih[cc]       + bhh[cc];           // r gate: both biases merge
            b_z[jg]  = bih[64 + cc]  + bhh[64 + cc];      // z gate: both biases merge
            b_in[jg] = bih[128 + cc];                     // n gate kept split
            b_hn[jg] = bhh[128 + cc];
        }
    };

    auto whh_frag = [&](int nt, int ks) -> v16bf {
        const uint4* p = (const uint4*)(whhf + (((nt * 2 + ks) * 32 + lane) << 3));
        uint4 a = p[0], b = p[1];
        return mk_frag(a.x, a.y, a.z, a.w, b.x, b.y, b.z, b.w);
    };
    // 16-bit A 16x32 fragment of h from the staging tile:
// lanes<16: K 0..15; lanes>=16: K 16..31  (+32*ks)
    auto h_frag = [&](int ks) -> v16bf {
        const uint4* p = (const uint4*)&stage[wave][laneN][32 * ks + 16 * laneH];
        uint4 a = p[0], b = p[1];
        return mk_frag(a.x, a.y, a.z, a.w, b.x, b.y, b.z, b.w);
    };

    auto gru_step = [&](const float* xrow) {
        // x A-fragment: row = lane (lanes<16), K 0..7 real, rest zero
        unsigned int xa0 = 0, xa1 = 0, xa2 = 0, xa3 = 0;
        if (xrow && lane < 16) {
            const float4* xp = (const float4*)(xrow + (size_t)lane * F_DIM);
            float4 p0 = xp[0], p1 = xp[1];
            xa0 = pk_bf16(p0.x, p0.y); xa1 = pk_bf16(p0.z, p0.w);
            xa2 = pk_bf16(p1.x, p1.y); xa3 = pk_bf16(p1.z, p1.w);
        }
        v16bf xf  = mk_frag(xa0, xa1, xa2, xa3, 0, 0, 0, 0);
        v16bf hf0 = h_frag(0);
        v16bf hf1 = h_frag(1);

        #pragma unroll
        for (int jg = 0; jg < 4; ++jg) {                  // 16 hidden units per pass
            v8f ar, az, agi, agh;
            #pragma unroll
            for (int v = 0; v < 8; ++v) {                 // bias as C-matrix init
                ar[v] = b_r[jg]; az[v] = b_z[jg];
                agi[v] = b_in[jg]; agh[v] = b_hn[jg];
            }
            v16bf wr = mk_frag(wih[jg    ][0], wih[jg    ][1], wih[jg    ][2], wih[jg    ][3], 0,0,0,0);
            v16bf wz = mk_frag(wih[jg + 4][0], wih[jg + 4][1], wih[jg + 4][2], wih[jg + 4][3], 0,0,0,0);
            v16bf wn = mk_frag(wih[jg + 8][0], wih[jg + 8][1], wih[jg + 8][2], wih[jg + 8][3], 0,0,0,0);

            ar  = wmma_bf16(xf,  wr,                 ar);
            ar  = wmma_bf16(hf0, whh_frag(jg,     0), ar);
            ar  = wmma_bf16(hf1, whh_frag(jg,     1), ar);
            az  = wmma_bf16(xf,  wz,                 az);
            az  = wmma_bf16(hf0, whh_frag(jg + 4, 0), az);
            az  = wmma_bf16(hf1, whh_frag(jg + 4, 1), az);
            agi = wmma_bf16(xf,  wn,                 agi);
            agh = wmma_bf16(hf0, whh_frag(jg + 8, 0), agh);
            agh = wmma_bf16(hf1, whh_frag(jg + 8, 1), agh);

            #pragma unroll
            for (int v = 0; v < 8; ++v) {
                float r  = fsigmoid(ar[v]);
                float z  = fsigmoid(az[v]);
                float n  = ftanh(agi[v] + r * agh[v]);
                float hn = (1.0f - z) * n + z * hstate[jg][v];
                hstate[jg][v] = hn;
                stage[wave][v + 8 * laneH][16 * jg + laneN] = f2bf(hn);
            }
        }
    };

    // ---- h0 = 0 -----------------------------------------------------------
    #pragma unroll
    for (int jg = 0; jg < 4; ++jg)
        #pragma unroll
        for (int v = 0; v < 8; ++v) hstate[jg][v] = 0.0f;
    for (int i = lane; i < ROWS_W * H_DIM; i += 32)
        (&stage[wave][0][0])[i] = 0;

    // ======================= encoder =======================
    fill_whh(eWhh);
    load_gate_consts(eWih, eBih, eBhh);
    #pragma unroll 1
    for (int t = 0; t < T_STEPS; ++t) {
        if (t + 1 < T_STEPS && lane < 16)
            __builtin_prefetch(inputs + ((size_t)(t + 1) * BATCH + b0 + lane) * F_DIM, 0, 0);
        gru_step(inputs + ((size_t)t * BATCH + b0) * F_DIM);
    }

    // ======================= decoder =======================
    fill_whh(dWhh);                                   // barriers inside
    load_gate_consts(dWih, dBih, dBhh);

    // out_W as a single (N padded 8->16) B-fragment pair, kept in registers
    unsigned int ow[2][8];
    #pragma unroll
    for (int ks = 0; ks < 2; ++ks)
        #pragma unroll
        for (int v = 0; v < 8; ++v) {
            unsigned int w = 0u;
            if (laneN < 8) {
                int k0 = 32 * ks + 16 * laneH + 2 * v;
                w = pk_bf16(oW[laneN * H_DIM + k0], oW[laneN * H_DIM + k0 + 1]);
            }
            ow[ks][v] = w;
        }
    float ob = (laneN < 8) ? oB[laneN] : 0.0f;

    #pragma unroll 1
    for (int t = 0; t < T_STEPS; ++t) {
        const float* xrow = (t == 0) ? (const float*)nullptr
                                     : target + ((size_t)(t - 1) * BATCH + b0) * F_DIM;
        if (t + 1 < T_STEPS && lane < 16)
            __builtin_prefetch(target + ((size_t)t * BATCH + b0 + lane) * F_DIM, 0, 0);
        gru_step(xrow);

        // y = h_new @ outW^T + outb
        v8f y;
        #pragma unroll
        for (int v = 0; v < 8; ++v) y[v] = ob;
        v16bf nh0 = h_frag(0), nh1 = h_frag(1);
        y = wmma_bf16(nh0, mk_frag(ow[0][0], ow[0][1], ow[0][2], ow[0][3],
                                   ow[0][4], ow[0][5], ow[0][6], ow[0][7]), y);
        y = wmma_bf16(nh1, mk_frag(ow[1][0], ow[1][1], ow[1][2], ow[1][3],
                                   ow[1][4], ow[1][5], ow[1][6], ow[1][7]), y);
        if (laneN < 8) {
            float* op = out + ((size_t)t * BATCH + b0 + 8 * laneH) * F_DIM + laneN;
            #pragma unroll
            for (int v = 0; v < 8; ++v) op[(size_t)v * F_DIM] = y[v];
        }
    }
}

extern "C" void kernel_launch(void* const* d_in, const int* in_sizes, int n_in,
                              void* d_out, int out_size, void* d_ws, size_t ws_size,
                              hipStream_t stream) {
    (void)in_sizes; (void)n_in; (void)out_size; (void)d_ws; (void)ws_size;
    const float* inputs = (const float*)d_in[0];
    const float* target = (const float*)d_in[1];
    const float* eWih   = (const float*)d_in[2];
    const float* eWhh   = (const float*)d_in[3];
    const float* eBih   = (const float*)d_in[4];
    const float* eBhh   = (const float*)d_in[5];
    const float* dWih   = (const float*)d_in[6];
    const float* dWhh   = (const float*)d_in[7];
    const float* dBih   = (const float*)d_in[8];
    const float* dBhh   = (const float*)d_in[9];
    const float* oW     = (const float*)d_in[10];
    const float* oB     = (const float*)d_in[11];
    float* out          = (float*)d_out;

    dim3 grid(BATCH / (WAVES * ROWS_W));   // 512 blocks
    dim3 block(WAVES * 32);                // 128 threads = 4 waves (wave32)
    seq2seq_gru_persistent<<<grid, block, 0, stream>>>(
        inputs, target, eWih, eWhh, eBih, eBhh,
        dWih, dWhh, dBih, dBhh, oW, oB, out);
}
